// SAGE_78795470012588
// MI455X (gfx1250) — compile-verified, matrix operand order checked by reading, not run on hardware
//
#include <hip/hip_runtime.h>
#include <hip/hip_bf16.h>
#include <math.h>

typedef __attribute__((ext_vector_type(16))) _Float16 v16h;
typedef __attribute__((ext_vector_type(8)))  _Float16 v8h;
typedef __attribute__((ext_vector_type(8)))  float    v8f;

#define NN 100000
#define EE 1600000
#define DD 128
#define CC 64

// ---------------- utility kernels ----------------

__global__ void k_zero(float* __restrict__ p, int n) {
  int i = blockIdx.x * blockDim.x + threadIdx.x;
  if (i < n) p[i] = 0.0f;
}

__global__ void k_cvt_f16(const float* __restrict__ s, _Float16* __restrict__ d, int n) {
  int i = blockIdx.x * blockDim.x + threadIdx.x;
  if (i < n) d[i] = (_Float16)s[i];
}

// ---------------- edge scatter (wave per edge) ----------------
// lane L moves feat[src][4L..4L+3] into agg[dst] via f32 atomics; lane0 bumps degree.
__global__ void k_scatter(const int* __restrict__ ei,
                          const float* __restrict__ feat,
                          float* __restrict__ agg,
                          float* __restrict__ deg,
                          int addDeg) {
  int e    = blockIdx.x * (blockDim.x >> 5) + (threadIdx.x >> 5);
  int lane = threadIdx.x & 31;
  if (e >= EE) return;
  int s = ei[e];
  int d = ei[EE + e];
  const float4 v = *(const float4*)(feat + (size_t)s * DD + lane * 4);
  float* a = agg + (size_t)d * DD + lane * 4;
  atomicAdd(a + 0, v.x);
  atomicAdd(a + 1, v.y);
  atomicAdd(a + 2, v.z);
  atomicAdd(a + 3, v.w);
  if (addDeg && lane == 0) atomicAdd(deg + d, 1.0f);
}

// ---------------- mean + f16 convert ----------------
__global__ void k_mean16(const float* __restrict__ agg, const float* __restrict__ deg,
                         _Float16* __restrict__ m16) {
  int i = blockIdx.x * blockDim.x + threadIdx.x;
  if (i >= NN * DD) return;
  int node = i >> 7;
  float dv = fmaxf(deg[node], 1.0f);
  m16[i] = (_Float16)(agg[i] / dv);
}

// ---------------- WMMA fragment loader ----------------
// 16-bit A-matrix 16x32 layout (ISA 7.12.2): lane<16 -> K in {kb..kb+7, kb+16..kb+23},
// lane>=16 -> +8.  B (KxN) mirrors A with lane = column.  Two b128 loads per fragment.
__device__ __forceinline__ v16h load_frag(const _Float16* __restrict__ base,
                                          int row0, int kbase, int lane) {
  int m    = lane & 15;
  int koff = (lane >> 4) << 3;  // 0 or 8
  const _Float16* p = base + (size_t)(row0 + m) * DD + kbase + koff;
  v8h lo = *(const v8h*)p;
  v8h hi = *(const v8h*)(p + 16);
  v16h a;
#pragma unroll
  for (int i = 0; i < 8; i++) { a[i] = lo[i]; a[i + 8] = hi[i]; }
  return a;
}

// ---------------- SAGE conv: out = mean@Wl^T + bl + x@Wr^T, then row L2-normalize ----------
// one wave = 16 node rows x 128 output features (8 N-tiles), K=128 in 4 chunks of 32.
__global__ void k_sage_wmma(const _Float16* __restrict__ Am,  // [N,128] mean, f16
                            const _Float16* __restrict__ Wl,  // [128,128] f16 row-major
                            const float*    __restrict__ bl,  // [128]
                            const _Float16* __restrict__ Ax,  // [N,128] root feats, f16
                            const _Float16* __restrict__ Wr,  // [128,128] f16
                            float* __restrict__ out) {        // [N,128] pre-BN, f32
  int wave = blockIdx.x * (blockDim.x >> 5) + (threadIdx.x >> 5);
  int lane = threadIdx.x & 31;
  const int ntiles = NN / 16;
  if (wave >= ntiles) return;
  int row0 = wave * 16;

  v8f acc[8];
#pragma unroll
  for (int t = 0; t < 8; t++) acc[t] = {};

#pragma unroll
  for (int kc = 0; kc < 4; kc++) {
    int kbase = kc * 32;
    v16h am = load_frag(Am, row0, kbase, lane);
    v16h ax = load_frag(Ax, row0, kbase, lane);
#pragma unroll
    for (int t = 0; t < 8; t++) {
      v16h bL = load_frag(Wl, t * 16, kbase, lane);
      acc[t] = __builtin_amdgcn_wmma_f32_16x16x32_f16(false, am, false, bL,
                                                      (short)0, acc[t], false, false);
      v16h bR = load_frag(Wr, t * 16, kbase, lane);
      acc[t] = __builtin_amdgcn_wmma_f32_16x16x32_f16(false, ax, false, bR,
                                                      (short)0, acc[t], false, false);
    }
  }

  int n = lane & 15;
#pragma unroll
  for (int t = 0; t < 8; t++) {
    float b = bl[t * 16 + n];
#pragma unroll
    for (int j = 0; j < 8; j++) acc[t][j] += b;
  }

  // per-row L2 norm: row j (+8 for upper half-wave) spread over 16 lanes x 8 tiles
  float rinv[8];
#pragma unroll
  for (int j = 0; j < 8; j++) {
    float s = 0.0f;
#pragma unroll
    for (int t = 0; t < 8; t++) s += acc[t][j] * acc[t][j];
#pragma unroll
    for (int msk = 1; msk < 16; msk <<= 1) s += __shfl_xor(s, msk, 32);
    rinv[j] = 1.0f / fmaxf(sqrtf(s), 1e-12f);
  }

  int mofs = (lane >> 4) << 3;  // rows 0-7 in lower half-wave, 8-15 in upper
#pragma unroll
  for (int t = 0; t < 8; t++) {
#pragma unroll
    for (int j = 0; j < 8; j++) {
      out[(size_t)(row0 + mofs + j) * DD + t * 16 + n] = acc[t][j] * rinv[j];
    }
  }
}

// ---------------- BatchNorm stats / finalize / apply ----------------
__global__ void k_bn_stats(const float* __restrict__ h,
                           float* __restrict__ sum, float* __restrict__ sumsq) {
  int col = threadIdx.x;  // 128 threads
  float s = 0.0f, s2 = 0.0f;
  for (int r = blockIdx.x; r < NN; r += gridDim.x) {
    float v = h[(size_t)r * DD + col];
    s += v; s2 += v * v;
  }
  atomicAdd(&sum[col], s);
  atomicAdd(&sumsq[col], s2);
}

__global__ void k_bn_final(const float* __restrict__ sum, const float* __restrict__ sumsq,
                           const float* __restrict__ g, const float* __restrict__ be,
                           float* __restrict__ scale, float* __restrict__ shift) {
  int c = threadIdx.x;  // 128 threads
  float mu  = sum[c] * (1.0f / (float)NN);
  float var = sumsq[c] * (1.0f / (float)NN) - mu * mu;
  float rs  = rsqrtf(var + 1e-5f);
  float sc  = g[c] * rs;
  scale[c] = sc;
  shift[c] = be[c] - mu * sc;
}

// relu(bn(h)) [+ residual]; emits f32 (optional) and f16 (for next WMMA A-operand)
__global__ void k_bn_apply(const float* __restrict__ h, const float* __restrict__ scale,
                           const float* __restrict__ shift, const float* __restrict__ res,
                           float* __restrict__ o32, _Float16* __restrict__ o16) {
  int i = blockIdx.x * blockDim.x + threadIdx.x;
  if (i >= NN * DD) return;
  int c = i & 127;
  float v = fmaxf(h[i] * scale[c] + shift[c], 0.0f);
  if (res) v += res[i];
  if (o32) o32[i] = v;
  o16[i] = (_Float16)v;
}

// ---------------- output head: logits = h@Wout^T + bout, then log_softmax ----------
__global__ void k_out_wmma(const _Float16* __restrict__ A,   // [N,128] f16
                           const _Float16* __restrict__ W,   // [64,128] f16
                           const float*    __restrict__ b,   // [64]
                           float* __restrict__ out) {        // [N,64]
  int wave = blockIdx.x * (blockDim.x >> 5) + (threadIdx.x >> 5);
  int lane = threadIdx.x & 31;
  const int ntiles = NN / 16;
  if (wave >= ntiles) return;
  int row0 = wave * 16;

  v8f acc[4];
#pragma unroll
  for (int t = 0; t < 4; t++) acc[t] = {};

#pragma unroll
  for (int kc = 0; kc < 4; kc++) {
    int kbase = kc * 32;
    v16h a = load_frag(A, row0, kbase, lane);
#pragma unroll
    for (int t = 0; t < 4; t++) {
      v16h bb = load_frag(W, t * 16, kbase, lane);
      acc[t] = __builtin_amdgcn_wmma_f32_16x16x32_f16(false, a, false, bb,
                                                      (short)0, acc[t], false, false);
    }
  }

  int n = lane & 15;
  float bias[4];
#pragma unroll
  for (int t = 0; t < 4; t++) bias[t] = b[t * 16 + n];

  float rmax[8];
#pragma unroll
  for (int j = 0; j < 8; j++) {
    float m = -3.402823e38f;
#pragma unroll
    for (int t = 0; t < 4; t++) m = fmaxf(m, acc[t][j] + bias[t]);
#pragma unroll
    for (int msk = 1; msk < 16; msk <<= 1) m = fmaxf(m, __shfl_xor(m, msk, 32));
    rmax[j] = m;
  }
  float rlse[8];
#pragma unroll
  for (int j = 0; j < 8; j++) {
    float s = 0.0f;
#pragma unroll
    for (int t = 0; t < 4; t++) s += expf(acc[t][j] + bias[t] - rmax[j]);
#pragma unroll
    for (int msk = 1; msk < 16; msk <<= 1) s += __shfl_xor(s, msk, 32);
    rlse[j] = logf(s);
  }

  int mofs = (lane >> 4) << 3;
#pragma unroll
  for (int t = 0; t < 4; t++) {
#pragma unroll
    for (int j = 0; j < 8; j++) {
      out[(size_t)(row0 + mofs + j) * CC + t * 16 + n] =
          acc[t][j] + bias[t] - rmax[j] - rlse[j];
    }
  }
}

// ---------------- host launcher ----------------
extern "C" void kernel_launch(void* const* d_in, const int* in_sizes, int n_in,
                              void* d_out, int out_size, void* d_ws, size_t ws_size,
                              hipStream_t stream) {
  const float* x    = (const float*)d_in[0];
  const int*   ei   = (const int*)d_in[1];   // [2,E] int32
  const float* W1l  = (const float*)d_in[2];
  const float* b1l  = (const float*)d_in[3];
  const float* W1r  = (const float*)d_in[4];
  const float* g1   = (const float*)d_in[5];
  const float* be1  = (const float*)d_in[6];
  const float* W2l  = (const float*)d_in[7];
  const float* b2l  = (const float*)d_in[8];
  const float* W2r  = (const float*)d_in[9];
  const float* g2   = (const float*)d_in[10];
  const float* be2  = (const float*)d_in[11];
  const float* Wout = (const float*)d_in[12];
  const float* bout = (const float*)d_in[13];
  float* out = (float*)d_out;

  // bump allocator over d_ws (256B-aligned slabs)
  char* base = (char*)d_ws;
  size_t cur = 0;
  auto alloc = [&](size_t bytes) -> void* {
    void* p = base + cur;
    cur = (cur + bytes + 255) & ~(size_t)255;
    return p;
  };
  float*    deg   = (float*)alloc((size_t)NN * 4);
  float*    agg   = (float*)alloc((size_t)NN * DD * 4);  // reused as pre-BN acts
  float*    h1    = (float*)alloc((size_t)NN * DD * 4);
  _Float16* x16   = (_Float16*)alloc((size_t)NN * DD * 2);  // reused as h1_16
  _Float16* m16   = (_Float16*)alloc((size_t)NN * DD * 2);  // reused as h2_16
  _Float16* w1l16 = (_Float16*)alloc(DD * DD * 2);
  _Float16* w1r16 = (_Float16*)alloc(DD * DD * 2);
  _Float16* w2l16 = (_Float16*)alloc(DD * DD * 2);
  _Float16* w2r16 = (_Float16*)alloc(DD * DD * 2);
  _Float16* wo16  = (_Float16*)alloc(CC * DD * 2);
  float* bnsum   = (float*)alloc(128 * 4);   // contiguous with bnsumsq (512B slabs)
  float* bnsumsq = (float*)alloc(128 * 4);
  float* bnscale = (float*)alloc(128 * 4);
  float* bnshift = (float*)alloc(128 * 4);

  const int T = 256;
  auto blks = [&](long long n) { return (unsigned)((n + T - 1) / T); };
  const long long ND = (long long)NN * DD;
  const unsigned wblocks = (NN / 16 + 7) / 8;   // 8 waves / block
  const unsigned eblocks = (EE + 7) / 8;        // wave per edge, 8 edges / block

  // f16 weight + feature staging
  k_cvt_f16<<<blks(DD * DD), T, 0, stream>>>(W1l, w1l16, DD * DD);
  k_cvt_f16<<<blks(DD * DD), T, 0, stream>>>(W1r, w1r16, DD * DD);
  k_cvt_f16<<<blks(DD * DD), T, 0, stream>>>(W2l, w2l16, DD * DD);
  k_cvt_f16<<<blks(DD * DD), T, 0, stream>>>(W2r, w2r16, DD * DD);
  k_cvt_f16<<<blks(CC * DD), T, 0, stream>>>(Wout, wo16, CC * DD);
  k_cvt_f16<<<blks(ND), T, 0, stream>>>(x, x16, (int)ND);

  // ---- layer 1 ----
  k_zero<<<blks(NN), T, 0, stream>>>(deg, NN);
  k_zero<<<blks(ND), T, 0, stream>>>(agg, (int)ND);
  k_scatter<<<eblocks, T, 0, stream>>>(ei, x, agg, deg, 1);
  k_mean16<<<blks(ND), T, 0, stream>>>(agg, deg, m16);
  k_sage_wmma<<<wblocks, T, 0, stream>>>(m16, w1l16, b1l, x16, w1r16, agg);
  k_zero<<<1, T, 0, stream>>>(bnsum, 256);  // bnsum + bnsumsq
  k_bn_stats<<<1024, 128, 0, stream>>>(agg, bnsum, bnsumsq);
  k_bn_final<<<1, 128, 0, stream>>>(bnsum, bnsumsq, g1, be1, bnscale, bnshift);
  k_bn_apply<<<blks(ND), T, 0, stream>>>(agg, bnscale, bnshift, nullptr, h1, x16);

  // ---- layer 2 (+ residual) ----
  k_zero<<<blks(ND), T, 0, stream>>>(agg, (int)ND);
  k_scatter<<<eblocks, T, 0, stream>>>(ei, h1, agg, deg, 0);
  k_mean16<<<blks(ND), T, 0, stream>>>(agg, deg, m16);
  k_sage_wmma<<<wblocks, T, 0, stream>>>(m16, w2l16, b2l, x16 /*h1_16*/, w2r16, agg);
  k_zero<<<1, T, 0, stream>>>(bnsum, 256);
  k_bn_stats<<<1024, 128, 0, stream>>>(agg, bnsum, bnsumsq);
  k_bn_final<<<1, 128, 0, stream>>>(bnsum, bnsumsq, g2, be2, bnscale, bnshift);
  k_bn_apply<<<blks(ND), T, 0, stream>>>(agg, bnscale, bnshift, h1, nullptr, m16 /*h2_16*/);

  // ---- output head + log_softmax ----
  k_out_wmma<<<wblocks, T, 0, stream>>>(m16, wo16, bout, out);
}
